// GATEncoder_67654324847546
// MI455X (gfx1250) — compile-verified, matrix operand order checked by reading, not run on hardware
//
#include <hip/hip_runtime.h>
#include <hip/hip_bf16.h>

// ---------------- problem constants (match reference) ----------------
#define Nn   50000
#define Ee   400000
#define Dd   256
#define H1f  32
#define HEADSn 4
#define H2f  64
#define DECn 512
#define Bn   4096
#define NEGs 0.2f

typedef __attribute__((ext_vector_type(16))) _Float16 v16h;
typedef __attribute__((ext_vector_type(8)))  _Float16 v8h;
typedef __attribute__((ext_vector_type(8)))  float    v8f;

// monotone float<->uint key for atomicMax on floats
__device__ __forceinline__ unsigned fkey(float f) {
  unsigned u = __float_as_uint(f);
  return (u & 0x80000000u) ? ~u : (u | 0x80000000u);
}
__device__ __forceinline__ float funkey(unsigned k) {
  return (k & 0x80000000u) ? __uint_as_float(k & 0x7FFFFFFFu) : __uint_as_float(~k);
}

// ---------------- WMMA GEMM: C[n,m] = A[n,k] @ B[k,m] (+bias)(+relu) ----------------
// f32 in/out, f16 multiply via v_wmma_f32_16x16x32_f16, f32 accumulate.
// Block = 4 waves = 64 rows x 64 cols. B panel (k x 64) is converted to f16 once
// per block into LDS, stored transposed bpT[col][k] (+8-half pad vs bank conflicts),
// so each lane's 16-half fragment = two contiguous 8-half runs -> 2x ds_load_b128.
// A side: per-lane rows, two contiguous 8-float K runs -> 4x global b128 loads.
// One A fragment per K-step feeds 4 independent WMMAs (no RAW hazard, XDL pipelining).
// n mult of 16, m mult of 64, k mult of 32.
template <bool BIAS, bool RELU>
__global__ void wmma_gemm_kernel(const float* __restrict__ A, const float* __restrict__ B,
                                 const float* __restrict__ bias, float* __restrict__ C,
                                 int n, int k, int m) {
  extern __shared__ _Float16 bpT[];      // [64][k+8]
  const int kpad = k + 8;
  const int cw = blockIdx.y;             // 64-wide column slab index
  const int col_base = cw * 64;

  // ---- cooperative B panel load: global f32 (coalesced) -> LDS f16 (transposed)
  for (int idx = threadIdx.x; idx < (k << 6); idx += blockDim.x) {
    int kq = idx >> 6, c = idx & 63;
    bpT[c * kpad + kq] = (_Float16)B[(size_t)kq * m + col_base + c];
  }
  __syncthreads();

  const int rt = blockIdx.x * 4 + (threadIdx.x >> 5);   // 16-row strip
  if (rt >= (n >> 4)) return;            // wave-uniform: EXEC stays all-ones
  const int lane = threadIdx.x & 31;
  const int Ml = lane & 15;
  const int hi8 = (lane & 16) ? 8 : 0;
  const size_t arow = (size_t)(rt * 16 + Ml) * (size_t)k;

  v8f acc0 = {0.f,0.f,0.f,0.f,0.f,0.f,0.f,0.f};
  v8f acc1 = acc0, acc2 = acc0, acc3 = acc0;

  for (int kk = 0; kk < k; kk += 32) {
    if (kk + 32 < k)
      __builtin_prefetch((const void*)(A + arow + kk + 32), 0, 1);  // global_prefetch_b8

    // A fragment: K runs {0..7}+hi8 and {16..23}+hi8 (ISA 7.12.2 16-bit layout)
    float4 q0 = *(const float4*)(A + arow + (size_t)(kk + hi8));
    float4 q1 = *(const float4*)(A + arow + (size_t)(kk + hi8 + 4));
    float4 q2 = *(const float4*)(A + arow + (size_t)(kk + 16 + hi8));
    float4 q3 = *(const float4*)(A + arow + (size_t)(kk + 16 + hi8 + 4));
    v16h a;
    a[0] = (_Float16)q0.x;  a[1] = (_Float16)q0.y;  a[2] = (_Float16)q0.z;  a[3] = (_Float16)q0.w;
    a[4] = (_Float16)q1.x;  a[5] = (_Float16)q1.y;  a[6] = (_Float16)q1.z;  a[7] = (_Float16)q1.w;
    a[8] = (_Float16)q2.x;  a[9] = (_Float16)q2.y;  a[10] = (_Float16)q2.z; a[11] = (_Float16)q2.w;
    a[12] = (_Float16)q3.x; a[13] = (_Float16)q3.y; a[14] = (_Float16)q3.z; a[15] = (_Float16)q3.w;

    // B fragments from LDS: 2x 16-byte ds loads per fragment
    v16h bfr[4];
#pragma unroll
    for (int c = 0; c < 4; ++c) {
      const _Float16* row = bpT + (size_t)(Ml + c * 16) * kpad + kk + hi8;
      v8h lo = *(const v8h*)row;
      v8h hh = *(const v8h*)(row + 16);
      bfr[c] = __builtin_shufflevector(lo, hh, 0,1,2,3,4,5,6,7,8,9,10,11,12,13,14,15);
    }
    acc0 = __builtin_amdgcn_wmma_f32_16x16x32_f16(false, a, false, bfr[0], (short)0, acc0, false, false);
    acc1 = __builtin_amdgcn_wmma_f32_16x16x32_f16(false, a, false, bfr[1], (short)0, acc1, false, false);
    acc2 = __builtin_amdgcn_wmma_f32_16x16x32_f16(false, a, false, bfr[2], (short)0, acc2, false, false);
    acc3 = __builtin_amdgcn_wmma_f32_16x16x32_f16(false, a, false, bfr[3], (short)0, acc3, false, false);
  }

  const int rbase = rt * 16 + hi8;       // C/D layout: lanes16-31 hold M=8..15
  v8f accs[4] = {acc0, acc1, acc2, acc3};
#pragma unroll
  for (int c = 0; c < 4; ++c) {
    int col = col_base + c * 16 + Ml;
    float bv = BIAS ? bias[col] : 0.f;
#pragma unroll
    for (int r = 0; r < 8; ++r) {
      float v = accs[c][r] + bv;
      if (RELU) v = v > 0.f ? v : 0.f;
      C[(size_t)(rbase + r) * m + col] = v;
    }
  }
}

// ---------------- attention scores a_s/a_d per (node, head) ----------------
__global__ void att_scores(const float* __restrict__ h, const float* __restrict__ aws,
                           const float* __restrict__ awd, float* __restrict__ as_,
                           float* __restrict__ ad_, int n, int H, int F) {
  int t = blockIdx.x * blockDim.x + threadIdx.x;
  if (t >= n * H) return;
  int i = t / H, hh = t - i * H;
  const float* row = h + (size_t)i * H * F + hh * F;
  float s = 0.f, d = 0.f;
  for (int f = 0; f < F; ++f) { s += row[f] * aws[hh * F + f]; d += row[f] * awd[hh * F + f]; }
  as_[t] = s; ad_[t] = d;
}

__global__ void init_smax(unsigned* __restrict__ km, float* __restrict__ ss, int cnt) {
  int t = blockIdx.x * blockDim.x + threadIdx.x;
  if (t < cnt) { km[t] = 0x007FFFFFu; /* fkey(-inf) */ ss[t] = 0.f; }
}

__global__ void edge_max(const int* __restrict__ ei, const float* __restrict__ as_,
                         const float* __restrict__ ad_, unsigned* __restrict__ km,
                         int E, int Np, int H) {
  int t = blockIdx.x * blockDim.x + threadIdx.x;
  int tot = (E + Np) * H;
  if (t >= tot) return;
  int e = t / H, hh = t - e * H;
  int s = e < E ? ei[e] : (e - E);
  int d = e < E ? ei[E + e] : (e - E);
  float x = as_[s * H + hh] + ad_[d * H + hh];
  x = x >= 0.f ? x : NEGs * x;
  atomicMax(&km[d * H + hh], fkey(x));
}

__global__ void decode_smax(unsigned* __restrict__ km, int cnt) {
  int t = blockIdx.x * blockDim.x + threadIdx.x;
  if (t >= cnt) return;
  float f = funkey(km[t]);
  ((float*)km)[t] = f;
}

__global__ void edge_expsum(const int* __restrict__ ei, const float* __restrict__ as_,
                            const float* __restrict__ ad_, const float* __restrict__ smax,
                            float* __restrict__ ssum, int E, int Np, int H) {
  int t = blockIdx.x * blockDim.x + threadIdx.x;
  int tot = (E + Np) * H;
  if (t >= tot) return;
  int e = t / H, hh = t - e * H;
  int s = e < E ? ei[e] : (e - E);
  int d = e < E ? ei[E + e] : (e - E);
  float x = as_[s * H + hh] + ad_[d * H + hh];
  x = x >= 0.f ? x : NEGs * x;
  atomicAdd(&ssum[d * H + hh], expf(x - smax[d * H + hh]));
}

__global__ void edge_aggr(const int* __restrict__ ei, const float* __restrict__ as_,
                          const float* __restrict__ ad_, const float* __restrict__ smax,
                          const float* __restrict__ ssum, const float* __restrict__ hsrc,
                          float* __restrict__ agg, int E, int Np, int H, int F) {
  int t = blockIdx.x * blockDim.x + threadIdx.x;
  int tot = (E + Np) * H;
  if (t >= tot) return;
  int e = t / H, hh = t - e * H;
  int s = e < E ? ei[e] : (e - E);
  int d = e < E ? ei[E + e] : (e - E);
  float x = as_[s * H + hh] + ad_[d * H + hh];
  x = x >= 0.f ? x : NEGs * x;
  float alpha = expf(x - smax[d * H + hh]) / (ssum[d * H + hh] + 1e-16f);
  const float* hr = hsrc + (size_t)s * H * F + hh * F;
  float* ar = agg + (size_t)d * H * F + hh * F;
  for (int f = 0; f < F; ++f) atomicAdd(&ar[f], hr[f] * alpha);
}

__global__ void finalize_prelu(const float* agg, const float* __restrict__ bias,
                               const float* __restrict__ pw, float* out, int n, int HF) {
  int t = blockIdx.x * blockDim.x + threadIdx.x;
  if (t >= n * HF) return;
  int f = t % HF;
  float v = agg[t] + bias[f];
  out[t] = v >= 0.f ? v : pw[f] * v;
}

// ---------------- readout / discriminator / adversarial ----------------
__global__ void colsum64(const float* __restrict__ x, float* __restrict__ out, int n) {
  __shared__ float s[64];
  int t = threadIdx.x;
  if (t < 64) s[t] = 0.f;
  __syncthreads();
  long tot = (long)n * 64;
  for (long i = (long)blockIdx.x * blockDim.x + t; i < tot; i += (long)gridDim.x * blockDim.x)
    atomicAdd(&s[(int)(i & 63)], x[i]);
  __syncthreads();
  if (t < 64) atomicAdd(&out[t], s[t]);
}

__global__ void readout_k(const float* __restrict__ cso, const float* __restrict__ csa,
                          const float* __restrict__ mlpW, const float* __restrict__ mlpb,
                          float* __restrict__ hos, float* __restrict__ hosa, int n) {
  __shared__ float so[64], sa[64];
  int j = threadIdx.x;
  so[j] = 1.f / (1.f + expf(-cso[j] / (float)n));
  sa[j] = 1.f / (1.f + expf(-csa[j] / (float)n));
  __syncthreads();
  float ao = 0.f, aa = 0.f;
  for (int kq = 0; kq < 64; ++kq) { float w = mlpW[kq * 64 + j]; ao += so[kq] * w; aa += sa[kq] * w; }
  hos[j] = ao + mlpb[j];
  hosa[j] = aa + mlpb[j];
}

__global__ void wc_adv_k(const float* __restrict__ discW, const float* __restrict__ hos,
                         const float* __restrict__ hosa, const float* __restrict__ advW,
                         const float* __restrict__ advb, float* __restrict__ wcos,
                         float* __restrict__ wcosa, float* __restrict__ advw) {
  int i = threadIdx.x;
  float a = 0.f, b = 0.f, c = 0.f;
  for (int kq = 0; kq < 64; ++kq) {
    float w = discW[i * 64 + kq];
    a += w * hos[kq];
    b += w * hosa[kq];
    c += advW[i * 64 + kq];   // row-sum: sum_j (x@advW)_j == x @ rowsum(advW)
  }
  wcos[i] = a; wcosa[i] = b; advw[i] = c;
  if (i == 0) { float s = 0.f; for (int j = 0; j < 64; ++j) s += advb[j]; advw[64] = s; }
}

__global__ void node_scores(const float* __restrict__ x2o, const float* __restrict__ x2a,
                            const float* __restrict__ wcos, const float* __restrict__ wcosa,
                            const float* __restrict__ advw, const float* __restrict__ discb,
                            float* __restrict__ retos, float* __restrict__ retosa,
                            float* __restrict__ logits, int n) {
  int i = blockIdx.x * blockDim.x + threadIdx.x;
  if (i >= n) return;
  const float* ro = x2o + (size_t)i * 64;
  const float* ra = x2a + (size_t)i * 64;
  float s1 = 0.f, s2 = 0.f, s3 = 0.f, s4 = 0.f, l1 = 0.f, l2 = 0.f;
  for (int kq = 0; kq < 64; ++kq) {
    float vo = ro[kq], va = ra[kq];
    s1 += vo * wcos[kq];  s2 += va * wcos[kq];
    s3 += va * wcosa[kq]; s4 += vo * wcosa[kq];
    l1 += vo * advw[kq];  l2 += va * advw[kq];
  }
  float db = discb[0];
  retos[i * 2] = s1 + db;  retos[i * 2 + 1] = s2 + db;
  retosa[i * 2] = s3 + db; retosa[i * 2 + 1] = s4 + db;
  logits[i] = l1 + advw[64];
  logits[n + i] = l2 + advw[64];
}

// ---------------- entity decoder ----------------
__global__ void ent_gather(const float* __restrict__ x2o, const int* __restrict__ idx,
                           float* __restrict__ ent, int Bcnt) {
  int t = blockIdx.x * blockDim.x + threadIdx.x;
  if (t >= Bcnt * 128) return;
  int b = t >> 7, f = t & 127;
  int node = (f < 64) ? idx[b] : idx[Bcnt + b];
  ent[t] = x2o[(size_t)node * 64 + (f & 63)];
}

__global__ void ent_final(const float* __restrict__ fh, const float* __restrict__ fW2,
                          const float* __restrict__ fb2, const float* __restrict__ fW3,
                          const float* __restrict__ fb3, float* __restrict__ logk,
                          float* __restrict__ log1k, int Bcnt) {
  int b = blockIdx.x * blockDim.x + threadIdx.x;
  if (b >= Bcnt) return;
  const float* r = fh + (size_t)b * 512;
  float a0 = 0.f, a1 = 0.f, c0 = 0.f, c1 = 0.f;
  for (int j = 0; j < 512; ++j) {
    float v = r[j];
    a0 += v * fW2[j * 2];  a1 += v * fW2[j * 2 + 1];
    c0 += v * fW3[j * 2];  c1 += v * fW3[j * 2 + 1];
  }
  logk[b * 2] = a0 + fb2[0];  logk[b * 2 + 1] = a1 + fb2[1];
  log1k[b * 2] = c0 + fb3[0]; log1k[b * 2 + 1] = c1 + fb3[1];
}

// ---------------- launcher ----------------
static inline int cdiv(long a, long b) { return (int)((a + b - 1) / b); }
static inline size_t gemm_smem(int k) { return (size_t)64 * (k + 8) * sizeof(_Float16); }

extern "C" void kernel_launch(void* const* d_in, const int* in_sizes, int n_in,
                              void* d_out, int out_size, void* d_ws, size_t ws_size,
                              hipStream_t stream) {
  const float* x_o  = (const float*)d_in[0];
  const float* x_a  = (const float*)d_in[1];
  const int*   ei   = (const int*)d_in[2];
  const int*   idx  = (const int*)d_in[3];
  const float* W1   = (const float*)d_in[4];
  const float* as1w = (const float*)d_in[5];
  const float* ad1w = (const float*)d_in[6];
  const float* b1   = (const float*)d_in[7];
  const float* p1   = (const float*)d_in[8];
  const float* W2   = (const float*)d_in[9];
  const float* as2w = (const float*)d_in[10];
  const float* ad2w = (const float*)d_in[11];
  const float* b2   = (const float*)d_in[12];
  const float* p2   = (const float*)d_in[13];
  const float* mlpW = (const float*)d_in[14];
  const float* mlpb = (const float*)d_in[15];
  const float* discW= (const float*)d_in[16];
  const float* discb= (const float*)d_in[17];
  const float* advW = (const float*)d_in[18];
  const float* advb = (const float*)d_in[19];
  const float* fW1  = (const float*)d_in[20];
  const float* fb1  = (const float*)d_in[21];
  const float* fW2  = (const float*)d_in[22];
  const float* fb2  = (const float*)d_in[23];
  const float* fW3  = (const float*)d_in[24];
  const float* fb3  = (const float*)d_in[25];

  float* ws = (float*)d_ws;
  size_t o = 0;
  float* buf_h   = ws + o; o += (size_t)Nn * 128;  // h1; reused for h2, then ent+fh
  float* buf_ax  = ws + o; o += (size_t)Nn * 128;  // agg1, then x1 in-place
  float* agg2    = ws + o; o += (size_t)Nn * 64;
  float* x2a     = ws + o; o += (size_t)Nn * 64;
  float* as1     = ws + o; o += (size_t)Nn * 4;
  float* ad1     = ws + o; o += (size_t)Nn * 4;
  float* sm1     = ws + o; o += (size_t)Nn * 4;    // uint keys, then floats in-place
  float* ss1     = ws + o; o += (size_t)Nn * 4;
  float* as2     = ws + o; o += Nn;
  float* ad2     = ws + o; o += Nn;
  float* sm2     = ws + o; o += Nn;
  float* ss2     = ws + o; o += Nn;
  float* cs_o    = ws + o; o += 64;
  float* cs_a    = ws + o; o += 64;
  float* hos     = ws + o; o += 64;
  float* hosa    = ws + o; o += 64;
  float* wcos    = ws + o; o += 64;
  float* wcosa   = ws + o; o += 64;
  float* advw    = ws + o; o += 65;
  float* ent = buf_h;                       // B*128, alias (free after both graphs)
  float* fh  = buf_h + (size_t)Bn * 128;    // B*512, alias

  float* out_log    = (float*)d_out + 0;
  float* out_retos  = (float*)d_out + 8192;
  float* out_retosa = (float*)d_out + 108192;
  float* out_x2o    = (float*)d_out + 208192;
  float* out_logits = (float*)d_out + 3408192;
  float* out_log1   = (float*)d_out + 3508192;

  const int EP = Ee + Nn;  // self-loops appended

  for (int g = 0; g < 2; ++g) {
    const float* xin = g == 0 ? x_o : x_a;
    float* x2 = g == 0 ? out_x2o : x2a;
    float* cs = g == 0 ? cs_o : cs_a;

    // Layer 1: h1 = x @ W1   (N x 256 @ 256 x 128)
    { dim3 grid(cdiv(Nn / 16, 4), 128 / 64);
      wmma_gemm_kernel<false, false><<<grid, 128, gemm_smem(Dd), stream>>>(xin, W1, nullptr, buf_h, Nn, Dd, 128); }
    att_scores<<<cdiv((long)Nn * 4, 256), 256, 0, stream>>>(buf_h, as1w, ad1w, as1, ad1, Nn, 4, 32);
    init_smax<<<cdiv((long)Nn * 4, 256), 256, 0, stream>>>((unsigned*)sm1, ss1, Nn * 4);
    hipMemsetAsync(buf_ax, 0, (size_t)Nn * 128 * sizeof(float), stream);
    edge_max<<<cdiv((long)EP * 4, 256), 256, 0, stream>>>(ei, as1, ad1, (unsigned*)sm1, Ee, Nn, 4);
    decode_smax<<<cdiv((long)Nn * 4, 256), 256, 0, stream>>>((unsigned*)sm1, Nn * 4);
    edge_expsum<<<cdiv((long)EP * 4, 256), 256, 0, stream>>>(ei, as1, ad1, sm1, ss1, Ee, Nn, 4);
    edge_aggr<<<cdiv((long)EP * 4, 256), 256, 0, stream>>>(ei, as1, ad1, sm1, ss1, buf_h, buf_ax, Ee, Nn, 4, 32);
    finalize_prelu<<<cdiv((long)Nn * 128, 256), 256, 0, stream>>>(buf_ax, b1, p1, buf_ax, Nn, 128);

    // Layer 2: h2 = x1 @ W2   (N x 128 @ 128 x 64)
    { dim3 grid(cdiv(Nn / 16, 4), 64 / 64);
      wmma_gemm_kernel<false, false><<<grid, 128, gemm_smem(128), stream>>>(buf_ax, W2, nullptr, buf_h, Nn, 128, 64); }
    att_scores<<<cdiv((long)Nn, 256), 256, 0, stream>>>(buf_h, as2w, ad2w, as2, ad2, Nn, 1, 64);
    init_smax<<<cdiv((long)Nn, 256), 256, 0, stream>>>((unsigned*)sm2, ss2, Nn);
    hipMemsetAsync(agg2, 0, (size_t)Nn * 64 * sizeof(float), stream);
    edge_max<<<cdiv((long)EP, 256), 256, 0, stream>>>(ei, as2, ad2, (unsigned*)sm2, Ee, Nn, 1);
    decode_smax<<<cdiv((long)Nn, 256), 256, 0, stream>>>((unsigned*)sm2, Nn);
    edge_expsum<<<cdiv((long)EP, 256), 256, 0, stream>>>(ei, as2, ad2, sm2, ss2, Ee, Nn, 1);
    edge_aggr<<<cdiv((long)EP, 256), 256, 0, stream>>>(ei, as2, ad2, sm2, ss2, buf_h, agg2, Ee, Nn, 1, 64);
    finalize_prelu<<<cdiv((long)Nn * 64, 256), 256, 0, stream>>>(agg2, b2, p2, x2, Nn, 64);

    hipMemsetAsync(cs, 0, 64 * sizeof(float), stream);
    colsum64<<<256, 256, 0, stream>>>(x2, cs, Nn);
  }

  readout_k<<<1, 64, 0, stream>>>(cs_o, cs_a, mlpW, mlpb, hos, hosa, Nn);
  wc_adv_k<<<1, 64, 0, stream>>>(discW, hos, hosa, advW, advb, wcos, wcosa, advw);
  node_scores<<<cdiv((long)Nn, 256), 256, 0, stream>>>(out_x2o, x2a, wcos, wcosa, advw, discb,
                                                       out_retos, out_retosa, out_logits, Nn);

  ent_gather<<<cdiv((long)Bn * 128, 256), 256, 0, stream>>>(out_x2o, idx, ent, Bn);
  { dim3 grid(cdiv(Bn / 16, 4), 512 / 64);
    wmma_gemm_kernel<true, true><<<grid, 128, gemm_smem(128), stream>>>(ent, fW1, fb1, fh, Bn, 128, 512); }
  ent_final<<<cdiv((long)Bn, 256), 256, 0, stream>>>(fh, fW2, fb2, fW3, fb3, out_log, out_log1, Bn);
}
